// GCN_3_52664888983661
// MI455X (gfx1250) — matrix operand order repaired and checked by reading, not compiled, not run on hardware
//
#include <hip/hip_runtime.h>
#include <math.h>

typedef __attribute__((ext_vector_type(16))) _Float16 v16h;
typedef __attribute__((ext_vector_type(8)))  _Float16 v8h;
typedef __attribute__((ext_vector_type(8)))  float    v8f;

static constexpr int kNodes = 100000;
static constexpr int kEdges = 1600000;
static constexpr int kFeat  = 256;
static constexpr int kHid   = 128;
static constexpr int kCls   = 40;
static constexpr int kClsP  = 48;   // padded class dim (multiple of 16)

// ---------------- f32 -> f16 conversion ----------------
__global__ void cvt_f16_kernel(const float* __restrict__ src, _Float16* __restrict__ dst, int n) {
  int i = blockIdx.x * blockDim.x + threadIdx.x;
  if (i < n) dst[i] = (_Float16)src[i];
}

// zero-padded conversion for W3 (rows x cols -> rows x colsPad)
__global__ void cvt_f16_pad_kernel(const float* __restrict__ src, _Float16* __restrict__ dst,
                                   int rows, int cols, int colsPad) {
  int i = blockIdx.x * blockDim.x + threadIdx.x;
  if (i >= rows * colsPad) return;
  int r = i / colsPad, c = i - r * colsPad;
  dst[i] = (c < cols) ? (_Float16)src[r * cols + c] : (_Float16)0.0f;
}

// ---------------- agg init: agg[i, f] = bias[f] ----------------
__global__ void bias_init_kernel(float* __restrict__ agg, const float* __restrict__ b,
                                 int n, int F) {
  int i = blockIdx.x * blockDim.x + threadIdx.x;
  if (i < n) agg[i] = b[i % F];
}

// ---------------- WMMA GEMM, register-blocked over N ----------------
// C[M x Npad] = A[M x K] * B[K x Npad]; A,B f16 row-major, C f32 (ldc = Npad).
// One wave computes a 16 x (NT*16) strip: one shared A fragment, NT accumulators,
// NT back-to-back WMMAs per 32-wide k-step. K % 32 == 0, Npad == NT*16, M % 16 == 0.
template <int NT>
__global__ void __launch_bounds__(256)
wmma_gemm_kernel(const _Float16* __restrict__ A, const _Float16* __restrict__ B,
                 float* __restrict__ C, int M, int K) {
  const int Npad = NT * 16;
  int tM   = (blockIdx.x * blockDim.x + threadIdx.x) >> 5;
  int lane = threadIdx.x & 31;
  if (tM * 16 >= M) return;                 // wave-uniform exit (EXEC all-1s for WMMA)

  int m15 = lane & 15;
  int h   = lane >> 4;                      // K-half select for A fragment
  const _Float16* Arow = A + (size_t)(tM * 16 + m15) * K;

  v8f acc[NT];
#pragma unroll
  for (int t = 0; t < NT; ++t) acc[t] = (v8f){};

  for (int k = 0; k < K; k += 32) {
    // A fragment: elements 0..7 = K[k+8h .. +7], 8..15 = K[k+16+8h .. +7]
    v8h alo = *(const v8h*)(Arow + k + 8 * h);
    v8h ahi = *(const v8h*)(Arow + k + 16 + 8 * h);
    v16h a;
#pragma unroll
    for (int e = 0; e < 8; ++e) { a[e] = alo[e]; a[e + 8] = ahi[e]; }
    // B fragments: lane l holds column N=t*16+(lane&15), K in same order as A
    const _Float16* Brow = B + (size_t)(k + 8 * h) * Npad + m15;
#pragma unroll
    for (int t = 0; t < NT; ++t) {
      v16h b;
#pragma unroll
      for (int e = 0; e < 8; ++e) { b[e] = Brow[(size_t)e * Npad + t * 16]; b[e + 8] = Brow[(size_t)(16 + e) * Npad + t * 16]; }
      acc[t] = __builtin_amdgcn_wmma_f32_16x16x32_f16(false, a, false, b, (short)0,
                                                      acc[t], false, false);
    }
  }
  // C: VGPR r -> M = tM*16 + r + 8*h, N = t*16 + (lane&15)
#pragma unroll
  for (int t = 0; t < NT; ++t) {
    float* Cp = C + (size_t)(tM * 16 + 8 * h) * Npad + t * 16 + m15;
#pragma unroll
    for (int r = 0; r < 8; ++r) Cp[(size_t)r * Npad] = acc[t][r];
  }
}

// ---------------- ReLU + f16 convert ----------------
__global__ void relu_cvt_kernel(const float* __restrict__ agg, _Float16* __restrict__ hh, int n) {
  int i = blockIdx.x * blockDim.x + threadIdx.x;
  if (i < n) { float v = agg[i]; hh[i] = (_Float16)(v > 0.0f ? v : 0.0f); }
}

// ---------------- edge scatter, F=128: one float4 per thread ----------------
__global__ void scatter128_kernel(const float* __restrict__ support, const int* __restrict__ src,
                                  const int* __restrict__ dst, const float* __restrict__ w,
                                  float* __restrict__ agg) {
  int idx = blockIdx.x * blockDim.x + threadIdx.x;   // kEdges * 32 threads
  int e = idx >> 5;
  if (e >= kEdges) return;
  int q = (idx & 31) << 2;
  float ww = w[e];
  const float4 s = *(const float4*)(support + (size_t)src[e] * kHid + q);
  float* ap = agg + (size_t)dst[e] * kHid + q;
  atomicAdd(ap + 0, ww * s.x);
  atomicAdd(ap + 1, ww * s.y);
  atomicAdd(ap + 2, ww * s.z);
  atomicAdd(ap + 3, ww * s.w);
}

// ---------------- edge scatter, F=40 (support padded to 48) ----------------
__global__ void scatter40_kernel(const float* __restrict__ support, const int* __restrict__ src,
                                 const int* __restrict__ dst, const float* __restrict__ w,
                                 float* __restrict__ agg) {
  int idx = blockIdx.x * blockDim.x + threadIdx.x;   // kEdges * 40 threads
  int e = idx / kCls;
  if (e >= kEdges) return;
  int f = idx - e * kCls;
  float ww = w[e];
  float s = support[(size_t)src[e] * kClsP + f];
  atomicAdd(agg + (size_t)dst[e] * kCls + f, ww * s);
}

// ---------------- log_softmax over 40 classes, one wave per node, in-place ----------------
__global__ void logsoftmax_kernel(float* __restrict__ out) {
  int wave = (blockIdx.x * blockDim.x + threadIdx.x) >> 5;
  int lane = threadIdx.x & 31;
  if (wave >= kNodes) return;
  float* row = out + (size_t)wave * kCls;
  float v0 = row[lane];
  float v1 = (lane < 8) ? row[32 + lane] : -3.4e38f;
  float m = fmaxf(v0, v1);
#pragma unroll
  for (int off = 16; off > 0; off >>= 1) m = fmaxf(m, __shfl_xor(m, off, 32));
  float s = expf(v0 - m) + ((lane < 8) ? expf(v1 - m) : 0.0f);
#pragma unroll
  for (int off = 16; off > 0; off >>= 1) s += __shfl_xor(s, off, 32);
  float lse = m + logf(s);
  row[lane] = v0 - lse;
  if (lane < 8) row[32 + lane] = v1 - lse;
}

extern "C" void kernel_launch(void* const* d_in, const int* in_sizes, int n_in,
                              void* d_out, int out_size, void* d_ws, size_t ws_size,
                              hipStream_t stream) {
  (void)in_sizes; (void)n_in; (void)out_size; (void)ws_size;
  const float* x    = (const float*)d_in[0];
  const int*   esrc = (const int*)d_in[1];
  const int*   edst = (const int*)d_in[2];
  const float* ev   = (const float*)d_in[3];
  const float* W1   = (const float*)d_in[4];
  const float* b1   = (const float*)d_in[5];
  const float* W2   = (const float*)d_in[6];
  const float* b2   = (const float*)d_in[7];
  const float* W3   = (const float*)d_in[8];
  const float* b3   = (const float*)d_in[9];
  float* out = (float*)d_out;

  // Workspace carve (256B aligned slices)
  char* p = (char*)d_ws;
  auto carve = [&](size_t bytes) { char* r = p; p += (bytes + 255) & ~(size_t)255; return r; };
  _Float16* xh      = (_Float16*)carve((size_t)kNodes * kFeat * 2);   // 51.2 MB
  _Float16* hh      = (_Float16*)carve((size_t)kNodes * kHid  * 2);   // 25.6 MB
  float*    support = (float*)   carve((size_t)kNodes * kHid  * 4);   // 51.2 MB (also 1e5x48 for L3)
  float*    agg     = (float*)   carve((size_t)kNodes * kHid  * 4);   // 51.2 MB
  _Float16* w1h     = (_Float16*)carve((size_t)kFeat * kHid  * 2);
  _Float16* w2h     = (_Float16*)carve((size_t)kHid  * kHid  * 2);
  _Float16* w3h     = (_Float16*)carve((size_t)kHid  * kClsP * 2);

  const int T = 256;
  auto blk = [](long long n, int t) { return (int)((n + t - 1) / t); };

  // f16 conversions
  cvt_f16_kernel<<<blk((long long)kNodes * kFeat, T), T, 0, stream>>>(x, xh, kNodes * kFeat);
  cvt_f16_kernel<<<blk(kFeat * kHid, T), T, 0, stream>>>(W1, w1h, kFeat * kHid);
  cvt_f16_kernel<<<blk(kHid * kHid, T), T, 0, stream>>>(W2, w2h, kHid * kHid);
  cvt_f16_pad_kernel<<<blk(kHid * kClsP, T), T, 0, stream>>>(W3, w3h, kHid, kCls, kClsP);

  const int mt = kNodes / 16;                       // 6250 16-row blocks; 1 wave each
  const int gemmBlocks = blk((long long)mt * 32, T);

  // ---- Layer 1 ----
  wmma_gemm_kernel<8><<<gemmBlocks, T, 0, stream>>>(xh, w1h, support, kNodes, kFeat);
  bias_init_kernel<<<blk((long long)kNodes * kHid, T), T, 0, stream>>>(agg, b1, kNodes * kHid, kHid);
  scatter128_kernel<<<blk((long long)kEdges * 32, T), T, 0, stream>>>(support, esrc, edst, ev, agg);
  relu_cvt_kernel<<<blk((long long)kNodes * kHid, T), T, 0, stream>>>(agg, hh, kNodes * kHid);

  // ---- Layer 2 ----
  wmma_gemm_kernel<8><<<gemmBlocks, T, 0, stream>>>(hh, w2h, support, kNodes, kHid);
  bias_init_kernel<<<blk((long long)kNodes * kHid, T), T, 0, stream>>>(agg, b2, kNodes * kHid, kHid);
  scatter128_kernel<<<blk((long long)kEdges * 32, T), T, 0, stream>>>(support, esrc, edst, ev, agg);
  relu_cvt_kernel<<<blk((long long)kNodes * kHid, T), T, 0, stream>>>(agg, hh, kNodes * kHid);

  // ---- Layer 3 (padded N=48, NT=3) ----
  wmma_gemm_kernel<3><<<gemmBlocks, T, 0, stream>>>(hh, w3h, support, kNodes, kHid);
  bias_init_kernel<<<blk((long long)kNodes * kCls, T), T, 0, stream>>>(out, b3, kNodes * kCls, kCls);
  scatter40_kernel<<<blk((long long)kEdges * kCls, T), T, 0, stream>>>(support, esrc, edst, ev, out);
  logsoftmax_kernel<<<blk((long long)kNodes * 32, T), T, 0, stream>>>(out);
}
